// MultiHeadAttention_37838661877847
// MI455X (gfx1250) — compile-verified, matrix operand order checked by reading, not run on hardware
//
#include <hip/hip_runtime.h>
#include <hip/hip_bf16.h>

// ---------------------------------------------------------------------------
// MI455X (gfx1250) multi-head attention.
// - All GEMMs via v_wmma_f32_16x16x32_f16 (fp32 accum), intermediates f16.
// - V is stored transposed [B,H,dk,S] by its projection so that BOTH the K
//   and V WMMA B-fragments are K-contiguous: all fragment loads in the
//   attention inner loop are 16B ds_load_b128 (no scalar gathers).
// - K/V tiles are DMA'd into LDS by the Tensor Data Mover (tensor_load_to_lds,
//   TENSORcnt), double-buffered: the DMA of block i+1 overlaps the WMMA +
//   softmax of block i. Tiles are shared by all 4 waves of the workgroup.
// - LDS row strides padded (K: 72 halfs, V/P: 40 halfs) -> conflict-free banks.
// - 1/sqrt(dk) folded into the Wq/bq f16 conversion.
// Workspace: Xq,Xk,Xv | Whq..Who | Qh,Kh [B,H,S,dk], Vt [B,H,dk,S] | Ah [B,S,D]
// ---------------------------------------------------------------------------

typedef __attribute__((ext_vector_type(16))) _Float16 v16h;
typedef __attribute__((ext_vector_type(8)))  _Float16 v8h;
typedef __attribute__((ext_vector_type(4)))  _Float16 v4h;
typedef __attribute__((ext_vector_type(8)))  float    v8f;
typedef __attribute__((ext_vector_type(4)))  unsigned int u32x4;
typedef __attribute__((ext_vector_type(8)))  int      i32x8;
typedef __attribute__((ext_vector_type(4)))  int      i32x4;

#define B_  4
#define S_  2048
#define D_  1024
#define H_  16
#define DK_ 64
#define M_  (B_ * S_)          // 8192 rows

#define TROWK_ 72   // K tile LDS row stride (halfs): 128B data + 16B pad
#define TROWV_ 40   // V tile LDS row stride (halfs): 64B data + 16B pad
#define PROW_  40   // P tile LDS row stride (halfs)
#define TLROW_ 20   // proj transpose-staging LDS row stride (halfs)

// Load a 32-K fragment whose K run is contiguous at p[0..31] (global or LDS).
// CDNA5 16-bit A/B 16x32 layout: per lane two contiguous 8-half runs at
// +8*half and +16+8*half.
__device__ __forceinline__ v16h load_frag_contig(const _Float16* p, int half) {
  v8h lo = *(const v8h*)(p + 8 * half);
  v8h hi = *(const v8h*)(p + 16 + 8 * half);
  v16h a;
#pragma unroll
  for (int j = 0; j < 8; ++j) { a[j] = lo[j]; a[j + 8] = hi[j]; }
  return a;
}

// ---------------------------------------------------------------------------
// TDM: DMA a 2D tile (tile_w x tile_h elements, source row stride `stride`)
// into LDS with 16B padding after every row (pad_code: row bytes = 8<<pad_code,
// pad_amount code 3 = 4 DWORDs). Descriptor per CDNA5 ISA 8.3/8.4.
// 6-arg builtin: (u32x4 g0, i32x8 g1, i32x4, i32x4, i32x8, i32 cpol)
__device__ __forceinline__ void tdm_load_tile(unsigned lds_addr, const _Float16* gsrc,
                                              int tile_w, int tile_h,
                                              int tensor_w, int tensor_h,
                                              int stride, int pad_code) {
  unsigned long long ga = (unsigned long long)(uintptr_t)gsrc;
  u32x4 g0;
  g0[0] = 1u;                                                // count=1 (valid D#)
  g0[1] = lds_addr;                                          // lds_addr (bytes)
  g0[2] = (unsigned)(ga & 0xFFFFFFFFu);                      // global_addr[31:0]
  g0[3] = (unsigned)((ga >> 32) & 0x01FFFFFFu) | (2u << 30); // addr[56:32]|type=2
  i32x8 g1;
  g1[0] = (1 << 16)                 // data_size = 1 -> 2 bytes
        | (1 << 20)                 // pad_enable
        | (pad_code << 22)          // pad_interval (DWORDs per row)
        | (3 << 25);                // pad_amount: 3 -> 4 DWORDs (16B)
  g1[1] = ((unsigned)tensor_w << 16);                               // tensor_dim0 lo
  g1[2] = ((unsigned)tensor_w >> 16) | ((unsigned)tensor_h << 16);  // dim0 hi|dim1 lo
  g1[3] = (((unsigned)tensor_h >> 16) & 0xFFFFu) | ((unsigned)tile_w << 16);
  g1[4] = tile_h;                   // tile_dim1 | tile_dim2=0
  g1[5] = stride;                   // tensor_dim0_stride (elements), low 32
  g1[6] = 0;
  g1[7] = 0;
  i32x4 z4 = (i32x4){0, 0, 0, 0};
  i32x8 z8 = (i32x8){0, 0, 0, 0, 0, 0, 0, 0};
  __builtin_amdgcn_tensor_load_to_lds(g0, g1, z4, z4, z8, 0);
}

// ---------------------------------------------------------------------------
__global__ __launch_bounds__(256)
void f32_to_f16_kernel(const float* __restrict__ src, _Float16* __restrict__ dst,
                       int n, float scale) {
  int i = blockIdx.x * blockDim.x + threadIdx.x;
  int stride = gridDim.x * blockDim.x;
  for (; i < n; i += stride) dst[i] = (_Float16)(src[i] * scale);
}

// ---------------------------------------------------------------------------
// y[m, n] = sum_k A[m,k] * W[n,k] + bscale*bias[n]   (nn.Linear: x @ W^T + b)
// One wave computes a 16x128 output tile (8 WMMA accumulators, 8x A reuse).
// MODE 0: fp32 out [M,D]   MODE 1: f16 [B,H,S,dk]   MODE 2: f16 [B,H,dk,S]
// (MODE 2 stages the tile through LDS so the transposed store is s-contiguous.)
template <int MODE>
__global__ __launch_bounds__(256)
void proj_kernel(const _Float16* __restrict__ A, const _Float16* __restrict__ W,
                 const float* __restrict__ bias, float bscale,
                 _Float16* __restrict__ outh, float* __restrict__ outf) {
  const int lane = threadIdx.x & 31;
  const int warp = threadIdx.x >> 5;
  const int half = lane >> 4;
  const int lr   = lane & 15;
  const int wid  = blockIdx.x * 8 + warp;      // 4096 wave-tiles total
  const int mtile = wid >> 3;                  // 512 row tiles of 16
  const int ntile = wid & 7;                   // 8 col tiles of 128
  const int m0 = mtile * 16;
  const int n0 = ntile * 128;

  v8f acc[8];
#pragma unroll
  for (int t = 0; t < 8; ++t) acc[t] = (v8f){};

  const _Float16* arow = A + (size_t)(m0 + lr) * D_;
  const _Float16* wp[8];
#pragma unroll
  for (int t = 0; t < 8; ++t) wp[t] = W + (size_t)(n0 + 16 * t + lr) * D_;

  for (int kb = 0; kb < D_; kb += 32) {
    v16h a = load_frag_contig(arow + kb, half);
    v16h bfr[8];
#pragma unroll
    for (int t = 0; t < 8; ++t) bfr[t] = load_frag_contig(wp[t] + kb, half);
#pragma unroll
    for (int t = 0; t < 8; ++t)
      acc[t] = __builtin_amdgcn_wmma_f32_16x16x32_f16(false, a, false, bfr[t],
                                                      (short)0, acc[t], false, false);
  }

  if constexpr (MODE != 2) {
#pragma unroll
    for (int t = 0; t < 8; ++t) {
      const int col = n0 + t * 16 + lr;
      const float bval = bias[col] * bscale;
#pragma unroll
      for (int r = 0; r < 8; ++r) {
        const int m = m0 + r + 8 * half;       // C layout: row = r + 8*half
        const float val = acc[t][r] + bval;
        if constexpr (MODE == 1) {
          const int b  = m >> 11;              // /S_
          const int s  = m & (S_ - 1);
          const int h  = col >> 6;             // /DK_
          const int dd = col & (DK_ - 1);
          outh[(((size_t)(b * H_ + h)) * S_ + s) * DK_ + dd] = (_Float16)val;
        } else {
          outf[(size_t)m * D_ + col] = val;
        }
      }
    }
  } else {
    // transpose the 16(s) x 128(col) tile through per-wave LDS, then store
    // s-contiguous rows of Vt[B,H,dk,S].
    __shared__ _Float16 tl[8][128 * TLROW_];
    _Float16* tw = &tl[warp][0];
#pragma unroll
    for (int t = 0; t < 8; ++t) {
      const int col = n0 + t * 16 + lr;
      const float bval = bias[col] * bscale;
#pragma unroll
      for (int r = 0; r < 8; ++r)
        tw[(t * 16 + lr) * TLROW_ + (r + 8 * half)] = (_Float16)(acc[t][r] + bval);
    }
    asm volatile("s_wait_dscnt 0" ::: "memory");   // wave-internal LDS visibility
    const int bq  = m0 >> 11;                  // batch (tile fits in one batch)
    const int s0r = m0 & (S_ - 1);             // sequence start
#pragma unroll
    for (int i = 0; i < 4; ++i) {
      const int cl  = i * 32 + lane;           // local col 0..127
      const int col = n0 + cl;
      const int hh  = col >> 6;
      const int dd  = col & (DK_ - 1);
      _Float16* dst = outh + (((size_t)(bq * H_ + hh)) * DK_ + dd) * S_ + s0r;
      *(v4h*)(dst +  0) = *(const v4h*)&tw[cl * TLROW_ +  0];
      *(v4h*)(dst +  4) = *(const v4h*)&tw[cl * TLROW_ +  4];
      *(v4h*)(dst +  8) = *(const v4h*)&tw[cl * TLROW_ +  8];
      *(v4h*)(dst + 12) = *(const v4h*)&tw[cl * TLROW_ + 12];
    }
  }
}

// ---------------------------------------------------------------------------
// Flash attention. One wave per (b, h, 16-row Q tile); the 4 waves of a block
// share (b,h) and consume the same K/V tiles, which the TDM DMAs into
// double-buffered LDS (warp0 -> K [32 keys x 64], warp1 -> V [64 dk x 32 keys])
// one 32-key block ahead.
__global__ __launch_bounds__(128)
void attention_kernel(const _Float16* __restrict__ Q, const _Float16* __restrict__ K,
                      const _Float16* __restrict__ Vt, _Float16* __restrict__ O) {
  __shared__ _Float16 Ktile[2][32 * TROWK_];   // 2 x 4.5 KB
  __shared__ _Float16 Vtile[2][64 * TROWV_];   // 2 x 5 KB
  __shared__ _Float16 plds[4][16 * PROW_];     // per-wave P tile, 4 x 1.25 KB

  const int lane = threadIdx.x & 31;
  const int warp = threadIdx.x >> 5;
  const int half = lane >> 4;
  const int lr   = lane & 15;
  const int wid  = blockIdx.x * 4 + warp;      // 8192 waves
  const int qt = wid & (S_ / 16 - 1);          // 0..127
  const int bh = wid >> 7;                     // 0..63 (uniform within block)
  const int h  = bh & (H_ - 1);
  const int b  = bh >> 4;

  const _Float16* Qb  = Q  + (size_t)bh * S_ * DK_ + (size_t)qt * 16 * DK_;
  const _Float16* Kb  = K  + (size_t)bh * S_ * DK_;
  const _Float16* Vtb = Vt + (size_t)bh * DK_ * S_;   // [dk][S] for this head

  // Q fragments: row = lr, dk=64 split into two 32-K fragments
  const v16h qa0 = load_frag_contig(Qb + lr * DK_ +  0, half);
  const v16h qa1 = load_frag_contig(Qb + lr * DK_ + 32, half);

  v8f o[4];
#pragma unroll
  for (int t = 0; t < 4; ++t) o[t] = (v8f){};
  float mi[8], li[8];
#pragma unroll
  for (int r = 0; r < 8; ++r) { mi[r] = -__builtin_inff(); li[r] = 0.0f; }

  // prologue: DMA tiles for block 0 into buffer 0
  if (warp == 0)   // K: 64 wide x 32 keys, row stride 64, 128B rows (code 4)
    tdm_load_tile((unsigned)(uintptr_t)&Ktile[0][0], Kb, DK_, 32, DK_, S_, DK_, 4);
  if (warp == 1)   // V: 32 keys wide x 64 dk rows, row stride S, 64B rows (code 3)
    tdm_load_tile((unsigned)(uintptr_t)&Vtile[0][0], Vtb, 32, DK_, S_, DK_, S_, 3);
  if (warp < 2)
    __builtin_amdgcn_s_wait_tensorcnt(0);
  __syncthreads();

  const int NBLK = S_ / 32;                    // 64 key blocks
  for (int ib = 0; ib < NBLK; ++ib) {
    const int cur = ib & 1;
    const int nxt = cur ^ 1;

    // issue next block's DMA now; it overlaps this block's WMMA + softmax
    if (ib + 1 < NBLK) {
      if (warp == 0)
        tdm_load_tile((unsigned)(uintptr_t)&Ktile[nxt][0],
                      Kb + (size_t)(ib + 1) * 32 * DK_, DK_, 32, DK_, S_, DK_, 4);
      if (warp == 1)
        tdm_load_tile((unsigned)(uintptr_t)&Vtile[nxt][0],
                      Vtb + (size_t)(ib + 1) * 32, 32, DK_, S_, DK_, S_, 3);
    }

    const _Float16* kt = &Ktile[cur][0];
    const _Float16* vt = &Vtile[cur][0];

    // --- K fragments from LDS (36-DW row stride: conflict-free) ---
    v16h kf[4];
    kf[0] = load_frag_contig(kt + (0  + lr) * TROWK_ +  0, half);
    kf[1] = load_frag_contig(kt + (0  + lr) * TROWK_ + 32, half);
    kf[2] = load_frag_contig(kt + (16 + lr) * TROWK_ +  0, half);
    kf[3] = load_frag_contig(kt + (16 + lr) * TROWK_ + 32, half);

    // --- V fragments from LDS, now K-contiguous too (20-DW row stride) ---
    v16h vf[4];
#pragma unroll
    for (int t4 = 0; t4 < 4; ++t4)
      vf[t4] = load_frag_contig(vt + (t4 * 16 + lr) * TROWV_, half);

    v8f s0 = (v8f){}, s1 = (v8f){};
    s0 = __builtin_amdgcn_wmma_f32_16x16x32_f16(false, qa0, false, kf[0], (short)0, s0, false, false);
    s0 = __builtin_amdgcn_wmma_f32_16x16x32_f16(false, qa1, false, kf[1], (short)0, s0, false, false);
    s1 = __builtin_amdgcn_wmma_f32_16x16x32_f16(false, qa0, false, kf[2], (short)0, s1, false, false);
    s1 = __builtin_amdgcn_wmma_f32_16x16x32_f16(false, qa1, false, kf[3], (short)0, s1, false, false);

    // --- online softmax per row r (row = r + 8*half, across 16 lanes) ---
#pragma unroll
    for (int r = 0; r < 8; ++r) {
      const float x0 = s0[r];                  // pre-scaled by 1/sqrt(dk)
      const float x1 = s1[r];
      float t = fmaxf(x0, x1);
#pragma unroll
      for (int mask = 1; mask < 16; mask <<= 1)
        t = fmaxf(t, __shfl_xor(t, mask, 32));
      const float mnew = fmaxf(mi[r], t);
      const float p0 = __expf(x0 - mnew);
      const float p1 = __expf(x1 - mnew);
      float sum = p0 + p1;
#pragma unroll
      for (int mask = 1; mask < 16; mask <<= 1)
        sum += __shfl_xor(sum, mask, 32);
      const float alpha = __expf(mi[r] - mnew);
      li[r] = li[r] * alpha + sum;
      mi[r] = mnew;
#pragma unroll
      for (int t4 = 0; t4 < 4; ++t4) o[t4][r] *= alpha;
      // spill P (C layout) to LDS row-major for re-layout into an A fragment
      plds[warp][(r + 8 * half) * PROW_ + lr]      = (_Float16)p0;
      plds[warp][(r + 8 * half) * PROW_ + 16 + lr] = (_Float16)p1;
    }
    asm volatile("s_wait_dscnt 0" ::: "memory");   // wave-internal LDS visibility

    // P as 16x32 A fragment (row = lr, K = key offset)
    const v16h pa = load_frag_contig(&plds[warp][lr * PROW_], half);

    // O += P (16x32) . V (32x64)
#pragma unroll
    for (int t4 = 0; t4 < 4; ++t4)
      o[t4] = __builtin_amdgcn_wmma_f32_16x16x32_f16(false, pa, false, vf[t4],
                                                     (short)0, o[t4], false, false);

    // next buffer must be complete and this buffer fully consumed
    if (ib + 1 < NBLK) {
      if (warp < 2)
        __builtin_amdgcn_s_wait_tensorcnt(0);
      __syncthreads();
    }
  }

  // normalize and write [B,S,D] f16
#pragma unroll
  for (int r = 0; r < 8; ++r) {
    const float inv = 1.0f / li[r];
    const int srow = qt * 16 + r + 8 * half;
    const size_t base = ((size_t)b * S_ + srow) * D_ + (size_t)h * DK_;
#pragma unroll
    for (int t4 = 0; t4 < 4; ++t4)
      O[base + t4 * 16 + lr] = (_Float16)(o[t4][r] * inv);
  }
}

// ---------------------------------------------------------------------------
extern "C" void kernel_launch(void* const* d_in, const int* in_sizes, int n_in,
                              void* d_out, int out_size, void* d_ws, size_t ws_size,
                              hipStream_t stream) {
  const float* q  = (const float*)d_in[0];
  const float* k  = (const float*)d_in[1];
  const float* v  = (const float*)d_in[2];
  const float* Wq = (const float*)d_in[3];
  const float* bq = (const float*)d_in[4];
  const float* Wk = (const float*)d_in[5];
  const float* bk = (const float*)d_in[6];
  const float* Wv = (const float*)d_in[7];
  const float* bv = (const float*)d_in[8];
  const float* Wo = (const float*)d_in[9];
  const float* bo = (const float*)d_in[10];

  const size_t MK = (size_t)M_ * D_;   // 8,388,608
  const size_t NK = (size_t)D_ * D_;   // 1,048,576
  const float attn_scale = 0.125f;     // 1/sqrt(dk), folded into Wq/bq

  _Float16* ws  = (_Float16*)d_ws;
  _Float16* Xq  = ws;
  _Float16* Xk  = Xq + MK;
  _Float16* Xv  = Xk + MK;
  _Float16* Whq = Xv + MK;
  _Float16* Whk = Whq + NK;
  _Float16* Whv = Whk + NK;
  _Float16* Who = Whv + NK;
  _Float16* Qh  = Who + NK;            // [B,H,S,dk]
  _Float16* Kh  = Qh + MK;             // [B,H,S,dk]
  _Float16* Vth = Kh + MK;             // [B,H,dk,S]  (transposed)
  _Float16* Ah  = Vth + MK;            // [B,S,D]

  // fp32 -> f16 conversions (Wq pre-scaled by 1/sqrt(dk))
  f32_to_f16_kernel<<<1024, 256, 0, stream>>>(q,  Xq,  (int)MK, 1.0f);
  f32_to_f16_kernel<<<1024, 256, 0, stream>>>(k,  Xk,  (int)MK, 1.0f);
  f32_to_f16_kernel<<<1024, 256, 0, stream>>>(v,  Xv,  (int)MK, 1.0f);
  f32_to_f16_kernel<<<512,  256, 0, stream>>>(Wq, Whq, (int)NK, attn_scale);
  f32_to_f16_kernel<<<512,  256, 0, stream>>>(Wk, Whk, (int)NK, 1.0f);
  f32_to_f16_kernel<<<512,  256, 0, stream>>>(Wv, Whv, (int)NK, 1.0f);
  f32_to_f16_kernel<<<512,  256, 0, stream>>>(Wo, Who, (int)NK, 1.0f);

  // projections (WMMA, 16x128 wave tiles)
  proj_kernel<1><<<512, 256, 0, stream>>>(Xq, Whq, bq, attn_scale, Qh, nullptr);
  proj_kernel<1><<<512, 256, 0, stream>>>(Xk, Whk, bk, 1.0f, Kh, nullptr);
  proj_kernel<2><<<512, 256, 0, stream>>>(Xv, Whv, bv, 1.0f, Vth, nullptr);

  // flash attention (WMMA + TDM double-buffered K/V tiles + online softmax)
  attention_kernel<<<2048, 128, 0, stream>>>(Qh, Kh, Vth, Ah);

  // output projection -> fp32 d_out
  proj_kernel<0><<<512, 256, 0, stream>>>(Ah, Who, bo, 1.0f, nullptr, (float*)d_out);
}
